// AttentionLSTM_69836168233535
// MI455X (gfx1250) — matrix operand order repaired and checked by reading, not compiled, not run on hardware
//
#include <hip/hip_runtime.h>
#include <hip/hip_bf16.h>

typedef unsigned short ushort_t;
typedef __attribute__((ext_vector_type(16))) __bf16 v16bf;
typedef __attribute__((ext_vector_type(8)))  __bf16 v8bf;
typedef __attribute__((ext_vector_type(8)))  float  v8f;

struct BfPair { v8bf lo, hi; };

#define N_BATCH 512
#define T_STEPS 128
#define D_IN    512
#define H_DIM   512
#define K_CAT   1536      // D + H + H
#define N_OUTC  2048      // 4*H
#define LDSROW  40        // 32 K elems + 8 pad (80B rows: conflict-free b128 reads)
#define KTILES  (K_CAT / 32)          // 48
#define BUF_ELEMS (128 * LDSROW)
#define BUF_BYTES (BUF_ELEMS * 2)

__device__ __forceinline__ ushort_t f2bf(float f) {
    unsigned u = __builtin_bit_cast(unsigned, f);
    u += 0x7FFFu + ((u >> 16) & 1u);       // round-to-nearest-even
    return (ushort_t)(u >> 16);
}

// ---------------------------------------------------------------------------
// One-time: WcatT[j][k] = bf16 of [Wx; Wh; Wattn][k][j]   (2048 x 1536, K-contig)
// ---------------------------------------------------------------------------
__global__ __launch_bounds__(256)
void prep_weights(const float* __restrict__ Wx, const float* __restrict__ Wh,
                  const float* __restrict__ Wattn, ushort_t* __restrict__ WcatT) {
    const int j = blockIdx.x;                       // 0..2047
    for (int k = threadIdx.x; k < K_CAT; k += 256) {
        float v;
        if (k < 512)       v = Wx[k * N_OUTC + j];
        else if (k < 1024) v = Wh[(k - 512) * N_OUTC + j];
        else               v = Wattn[(k - 1024) * N_OUTC + j];
        WcatT[j * K_CAT + k] = f2bf(v);
    }
}

// h0 = c0 = mean over the 16 spatial cells of A[n,h,:,:]
__global__ __launch_bounds__(256)
void init_state(const float* __restrict__ A, float* __restrict__ h_state,
                float* __restrict__ c_state) {
    const int idx = blockIdx.x * 256 + threadIdx.x;  // n*512 + h
    const float* ap = A + idx * 16;
    float s = 0.f;
    #pragma unroll
    for (int i = 0; i < 16; ++i) s += ap[i];
    s *= (1.f / 16.f);
    h_state[idx] = s;
    c_state[idx] = s;
}

// ---------------------------------------------------------------------------
// Per step: attention over 16 locations + build Act = bf16[x_t | h | attn]
// ---------------------------------------------------------------------------
__global__ __launch_bounds__(512)
void attn_step(const float* __restrict__ A, const float* __restrict__ x,
               const float* __restrict__ h_state, ushort_t* __restrict__ Act,
               int tstep) {
    __shared__ float dots[16];
    const int n = blockIdx.x;
    const int tid = threadIdx.x;
    const int p = tid >> 5, lane = tid & 31;
    const float* Ap = A + n * (H_DIM * 16);
    const float* hp = h_state + n * H_DIM;

    float partial = 0.f;
    for (int h = lane; h < H_DIM; h += 32)
        partial += Ap[h * 16 + p] * hp[h];
    #pragma unroll
    for (int off = 16; off > 0; off >>= 1)
        partial += __shfl_xor(partial, off, 32);
    if (lane == 0) dots[p] = partial * 0.04419417382415922f;   // 1/sqrt(512)
    __syncthreads();

    float d[16], mx = -1e30f;
    #pragma unroll
    for (int i = 0; i < 16; ++i) { d[i] = dots[i]; mx = fmaxf(mx, d[i]); }
    float sum = 0.f;
    #pragma unroll
    for (int i = 0; i < 16; ++i) { d[i] = __expf(d[i] - mx); sum += d[i]; }
    const float inv = 1.f / sum;

    const float* arow = Ap + tid * 16;
    float at = 0.f;
    #pragma unroll
    for (int i = 0; i < 16; ++i) at += d[i] * arow[i];
    at *= inv;

    ushort_t* actn = Act + n * K_CAT;
    actn[tid]                = f2bf(x[(n * T_STEPS + tstep) * D_IN + tid]);
    actn[D_IN + tid]         = f2bf(hp[tid]);
    actn[D_IN + H_DIM + tid] = f2bf(at);
}

// ---------------------------------------------------------------------------
// Hbar(512x2048) = Act(512x1536 bf16) @ Wcat(1536x2048 bf16), fp32 accum.
// 128x128 tile/block; double-buffered LDS fed by GLOBAL_LOAD_ASYNC_TO_LDS_B128.
// ---------------------------------------------------------------------------
__global__ __launch_bounds__(256)
void gemm_wmma(const ushort_t* __restrict__ Act, const ushort_t* __restrict__ WcatT,
               float* __restrict__ Hbar) {
    __shared__ __align__(16) ushort_t Ash[2][BUF_ELEMS];
    __shared__ __align__(16) ushort_t Bsh[2][BUF_ELEMS];

    const int tid  = threadIdx.x;
    const int lane = tid & 31;
    const int wave = tid >> 5;
    const int wm = wave & 1;            // M offset = wm*64
    const int wn = wave >> 1;           // N offset = wn*32
    const int m0 = blockIdx.x * 128;
    const int n0 = blockIdx.y * 128;
    const int laneM    = lane & 15;
    const int laneHalf = lane >> 4;

    // Staging: thread t owns row = t>>1 (0..127), 32B chunk pair (t&1)
    const int srow = tid >> 1;
    const int sseg = (tid & 1) * 2;     // segs {sseg, sseg+1}, 16B each

    const ushort_t* gA = Act   + (m0 + srow) * K_CAT + sseg * 8;
    const ushort_t* gB = WcatT + (n0 + srow) * K_CAT + sseg * 8;
    const unsigned  lA = (unsigned)(size_t)(&Ash[0][srow * LDSROW + sseg * 8]);
    const unsigned  lB = (unsigned)(size_t)(&Bsh[0][srow * LDSROW + sseg * 8]);

    v8f acc[4][2];
    #pragma unroll
    for (int mi = 0; mi < 4; ++mi)
        #pragma unroll
        for (int ni = 0; ni < 2; ++ni) {
            v8f z = {0.f, 0.f, 0.f, 0.f, 0.f, 0.f, 0.f, 0.f};
            acc[mi][ni] = z;
        }

    // async-stage one 8KB A tile + 8KB B tile (4 b128 per thread)
    auto issue = [&](int kt, int buf) {
        const ushort_t* a0 = gA + kt * 32;
        const ushort_t* b0 = gB + kt * 32;
        const unsigned  la = lA + (unsigned)buf * BUF_BYTES;
        const unsigned  lb = lB + (unsigned)buf * BUF_BYTES;
        asm volatile("global_load_async_to_lds_b128 %0, %1, off"
                     :: "v"(la), "v"(a0) : "memory");
        asm volatile("global_load_async_to_lds_b128 %0, %1, off offset:16"
                     :: "v"(la), "v"(a0) : "memory");
        asm volatile("global_load_async_to_lds_b128 %0, %1, off"
                     :: "v"(lb), "v"(b0) : "memory");
        asm volatile("global_load_async_to_lds_b128 %0, %1, off offset:16"
                     :: "v"(lb), "v"(b0) : "memory");
    };

    issue(0, 0);

    for (int kt = 0; kt < KTILES; ++kt) {
        const int buf = kt & 1;
        asm volatile("s_wait_asynccnt 0x0" ::: "memory");  // own tile-k loads landed
        __syncthreads();                                    // everyone's landed; prev reads done
        if (kt + 1 < KTILES) issue(kt + 1, buf ^ 1);        // prefetch overlaps compute

        const ushort_t* As = &Ash[buf][0];
        const ushort_t* Bs = &Bsh[buf][0];

        // load all 6 fragments into distinct registers first
        v16bf bfrag[2];
        #pragma unroll
        for (int ni = 0; ni < 2; ++ni) {
            const ushort_t* bp = Bs + (wn * 32 + ni * 16 + laneM) * LDSROW + laneHalf * 8;
            BfPair pr;
            pr.lo = *(const v8bf*)(bp);
            pr.hi = *(const v8bf*)(bp + 16);
            bfrag[ni] = __builtin_bit_cast(v16bf, pr);
        }
        v16bf afrag[4];
        #pragma unroll
        for (int mi = 0; mi < 4; ++mi) {
            const ushort_t* ap = As + (wm * 64 + mi * 16 + laneM) * LDSROW;
            BfPair pr;
            pr.lo = *(const v8bf*)(ap + laneHalf * 8);
            pr.hi = *(const v8bf*)(ap + 16 + laneHalf * 8);
            afrag[mi] = __builtin_bit_cast(v16bf, pr);
        }
        #pragma unroll
        for (int mi = 0; mi < 4; ++mi)
            #pragma unroll
            for (int ni = 0; ni < 2; ++ni)
                acc[mi][ni] = __builtin_amdgcn_wmma_f32_16x16x32_bf16(
                    false, afrag[mi], false, bfrag[ni], (short)0, acc[mi][ni], false, false);
    }

    // D layout: VGPR r -> M = r + 8*laneHalf, N = laneM
    #pragma unroll
    for (int mi = 0; mi < 4; ++mi)
        #pragma unroll
        for (int ni = 0; ni < 2; ++ni) {
            const int gm = m0 + wm * 64 + mi * 16 + 8 * laneHalf;
            const int gn = n0 + wn * 32 + ni * 16 + laneM;
            #pragma unroll
            for (int r = 0; r < 8; ++r)
                Hbar[(gm + r) * N_OUTC + gn] = acc[mi][ni][r];
        }
}

// ---------------------------------------------------------------------------
// gates + state update + output write
// ---------------------------------------------------------------------------
__global__ __launch_bounds__(256)
void gates_step(const float* __restrict__ Hbar, const float* __restrict__ b,
                float* __restrict__ h_state, float* __restrict__ c_state,
                float* __restrict__ out, int tstep) {
    const int idx = blockIdx.x * 256 + threadIdx.x;  // < 262144
    const int n = idx >> 9, j = idx & 511;
    const float* hb = Hbar + n * N_OUTC;
    const float vi = hb[j]        + b[j];
    const float vf = hb[512 + j]  + b[512 + j];
    const float vo = hb[1024 + j] + b[1024 + j];
    const float vg = hb[1536 + j] + b[1536 + j];
    const float ii = 1.f / (1.f + __expf(-vi));
    const float ff = 1.f / (1.f + __expf(-vf));
    const float oo = 1.f / (1.f + __expf(-vo));
    const float gg = tanhf(vg);
    const float c  = ff * c_state[idx] + ii * gg;
    const float hn = oo * tanhf(c);
    c_state[idx] = c;
    h_state[idx] = hn;
    out[n * (T_STEPS * H_DIM) + tstep * H_DIM + j] = hn;
}

// ---------------------------------------------------------------------------
extern "C" void kernel_launch(void* const* d_in, const int* in_sizes, int n_in,
                              void* d_out, int out_size, void* d_ws, size_t ws_size,
                              hipStream_t stream) {
    const float* x     = (const float*)d_in[0];
    const float* A     = (const float*)d_in[1];
    const float* Wx    = (const float*)d_in[2];
    const float* Wh    = (const float*)d_in[3];
    const float* Wattn = (const float*)d_in[4];
    const float* b     = (const float*)d_in[5];
    float* out = (float*)d_out;

    char* ws = (char*)d_ws;
    ushort_t* WcatT  = (ushort_t*)(ws);                          // 6.0 MB
    ushort_t* Act    = (ushort_t*)(ws + 6u  * 1024 * 1024);      // 1.5 MB
    float*   h_state = (float*)  (ws + 8u  * 1024 * 1024);       // 1.0 MB
    float*   c_state = (float*)  (ws + 9u  * 1024 * 1024);       // 1.0 MB
    float*   Hbar    = (float*)  (ws + 10u * 1024 * 1024);       // 4.0 MB

    prep_weights<<<N_OUTC, 256, 0, stream>>>(Wx, Wh, Wattn, WcatT);
    init_state<<<(N_BATCH * H_DIM) / 256, 256, 0, stream>>>(A, h_state, c_state);

    dim3 ggrid(N_BATCH / 128, N_OUTC / 128);
    for (int t = 0; t < T_STEPS; ++t) {
        attn_step<<<N_BATCH, 512, 0, stream>>>(A, x, h_state, Act, t);
        gemm_wmma<<<ggrid, 256, 0, stream>>>(Act, WcatT, Hbar);
        gates_step<<<(N_BATCH * H_DIM) / 256, 256, 0, stream>>>(Hbar, b, h_state, c_state, out, t);
    }
}
